// SpatialFusion_76639396429862
// MI455X (gfx1250) — matrix operand order repaired and checked by reading, not compiled, not run on hardware
//
#include <hip/hip_runtime.h>
#include <hip/hip_bf16.h>

typedef __attribute__((ext_vector_type(16))) _Float16 v16h;
typedef __attribute__((ext_vector_type(8)))  float    v8f;

// ---------------------------------------------------------------------------
// WMMA helpers (CDNA5 wave32 layouts, CDNA5 ISA §7.12.2)
// ---------------------------------------------------------------------------
__device__ __forceinline__ v8f wmma_f16(v16h a, v16h b, v8f c) {
  // 8 args: (neg_a, A, neg_b, B, c_mod, C, reuse_a, reuse_b)
  return __builtin_amdgcn_wmma_f32_16x16x32_f16(false, a, false, b, (short)0, c,
                                                false, false);
}

// A fragment: 16x32 f16 tile, row-major in LDS with row stride `ld` halves.
// lanes 0-15: row=lane,   VGPR0-3 K=0..7,  VGPR4-7 K=16..23
// lanes16-31: row=lane-16,VGPR0-3 K=8..15, VGPR4-7 K=24..31
__device__ __forceinline__ v16h ldsA(const _Float16* base, int lane, int ld) {
  const int r  = lane & 15;
  const int kb = (lane < 16) ? 0 : 8;
  const _Float16* row = base + r * ld;
  v16h a;
#pragma unroll
  for (int v = 0; v < 8; ++v) {
    const int k0 = ((v < 4) ? (2 * v) : (16 + 2 * (v - 4))) + kb;
    a[2 * v]     = row[k0];
    a[2 * v + 1] = row[k0 + 1];
  }
  return a;
}

// B fragment: 32x16 f16 tile. LDS holds W transposed: wT[n][k], stride ld.
// lanes 0-15: col=lane, K=0..7,16..23 ; lanes 16-31: col=lane-16, K=8..15,24..31
__device__ __forceinline__ v16h ldsB(const _Float16* baseT, int lane, int ld) {
  const int c = lane & 31 & 15;
  const int g = (lane & 31) >> 4;
  const _Float16* col = baseT + c * ld + g * 8;
  v16h b;
#pragma unroll
  for (int h = 0; h < 16; ++h) b[h] = col[(h < 8) ? h : (h + 8)];
  return b;
}

// ---------------------------------------------------------------------------
// Async global->LDS linear tile copy (GLOBAL_LOAD_ASYNC_TO_LDS_B128, GVS mode,
// tracked by ASYNCcnt; CDNA5 ISA). `src`/`dst` must be
// 16B aligned, BYTES a multiple of 16, and the tile fully in-range.
// ---------------------------------------------------------------------------
template <int BYTES>
__device__ __forceinline__ void async_stage(_Float16* dst, const _Float16* src,
                                            int tid) {
  // Low 32 bits of a flat shared-aperture pointer are the LDS byte address.
  unsigned l0 = (unsigned)(unsigned long long)(void*)dst;
#pragma unroll
  for (int c = 0; c < BYTES / 16 / 256; ++c) {
    unsigned off = (unsigned)(tid * 16 + c * 256 * 16);
    asm volatile("global_load_async_to_lds_b128 %0, %1, %2 offset:0"
                 :
                 : "v"(l0 + off), "v"(off), "s"(src)
                 : "memory");
  }
  // tail chunks (BYTES/16 not divisible by 256)
  {
    constexpr int done = (BYTES / 16 / 256) * 256 * 16;
    if (done + tid * 16 < BYTES) {
      unsigned off = (unsigned)(done + tid * 16);
      asm volatile("global_load_async_to_lds_b128 %0, %1, %2 offset:0"
                   :
                   : "v"(l0 + off), "v"(off), "s"(src)
                   : "memory");
    }
  }
}

__device__ __forceinline__ void async_wait0() {
  asm volatile("s_wait_asynccnt 0" ::: "memory");
}

__device__ __forceinline__ void atomicMaxF(float* addr, float val) {
  unsigned* ua = reinterpret_cast<unsigned*>(addr);
  unsigned cur = *ua;
  while (__uint_as_float(cur) < val) {
    unsigned prev = atomicCAS(ua, cur, __float_as_uint(val));
    if (prev == cur) break;
    cur = prev;
  }
}

// ---------------------------------------------------------------------------
// init: out = 0, m = -inf, s = 0
// ---------------------------------------------------------------------------
__global__ void init_bufs(float* __restrict__ out, float* __restrict__ m,
                          float* __restrict__ s, long n) {
  long i = (long)blockIdx.x * blockDim.x + threadIdx.x;
  if (i < n) { out[i] = 0.f; m[i] = -3.0e38f; s[i] = 0.f; }
}

// ---------------------------------------------------------------------------
// Build concat input x_cat (N x 192) in f16: [features | naip_c | uav_c]
// ---------------------------------------------------------------------------
__global__ void build_xcat(const float* __restrict__ pf, const float* __restrict__ pos,
                           const float* __restrict__ naip, const float* __restrict__ uav,
                           const float* __restrict__ nb, const float* __restrict__ ub,
                           const float* __restrict__ center, const float* __restrict__ scale,
                           _Float16* __restrict__ xcat, int Npts) {
  long i = (long)blockIdx.x * blockDim.x + threadIdx.x;
  long total = (long)Npts * 192;
  if (i >= total) return;
  int n = (int)(i / 192), c = (int)(i % 192);
  float v;
  if (c < 128) {
    v = pf[(size_t)n * 128 + c];
  } else {
    int which = (c - 128) >> 5;
    int dim   = (c - 128) & 31;
    const float* bb  = which ? ub : nb;
    const float* emb = which ? uav : naip;
    float minx = bb[0], miny = bb[1], maxx = bb[2], maxy = bb[3];
    float psx = (maxx - minx) * 0.25f, psy = (maxy - miny) * 0.25f;
    float cx = center[0], cy = center[1];
    float inv_s = 1.f / scale[0];
    float px = pos[(size_t)n * 3 + 0], py = pos[(size_t)n * 3 + 1];
    float wsum = 0.f, acc = 0.f;
#pragma unroll
    for (int p = 0; p < 16; ++p) {
      float gx = (minx + psx * (0.5f + (float)(p & 3)) - cx) * inv_s;
      float gy = (miny + psy * (0.5f + (float)(p >> 2)) - cy) * inv_s;
      float dx = px - gx, dy = py - gy;
      float d = sqrtf(dx * dx + dy * dy);
      float wp = (d <= 0.5f) ? __expf(-d * 10.f) : 0.f;
      wsum += wp;
      acc  += wp * emb[p * 32 + dim];
    }
    float valid = (wsum > 1e-6f) ? 1.f : 0.f;
    v = acc / (wsum + 1e-10f) * valid;
  }
  xcat[i] = (_Float16)v;
}

// ---------------------------------------------------------------------------
// Node GEMM: out(M x NC) = A(M x K, f16) @ W(K x NC, f32->f16) [+ bias]
// 256 threads = 8 waves; block tile 64 rows x NC cols; W staged in 64-K chunks.
// A tile staged via async global->LDS copies (full tiles) overlapping the
// weight transpose/convert done on the VALU.
// ---------------------------------------------------------------------------
template <int K_, int NC_, bool OUTH, bool BIAS>
__global__ __launch_bounds__(256) void gemm_node(
    const _Float16* __restrict__ A, const float* __restrict__ W,
    const float* __restrict__ bias, float* __restrict__ outF,
    _Float16* __restrict__ outH, int M) {
  extern __shared__ _Float16 sm[];
  _Float16* As = sm;             // [64][K_]
  _Float16* Ws = sm + 64 * K_;   // [NC_][64] transposed K-chunk
  const int tid = threadIdx.x;
  const int lane = tid & 31;
  const int w = tid >> 5;
  const int rt = w & 3;          // row tile 0..3 (16 rows each)
  const int cg = w >> 2;         // column group 0..1
  const int row0 = blockIdx.x * 64;
  const bool full = (row0 + 64 <= M);

  if (full) {
    async_stage<64 * K_ * 2>(As, A + (size_t)row0 * K_, tid);
  } else {
    for (int i = tid; i < 64 * K_ / 2; i += 256) {
      int r = (2 * i) / K_;
      int c = (2 * i) % K_;
      unsigned v = 0;
      if (row0 + r < M)
        v = *reinterpret_cast<const unsigned*>(A + (size_t)(row0 + r) * K_ + c);
      *reinterpret_cast<unsigned*>(As + r * K_ + c) = v;
    }
  }

  constexpr int PER = NC_ / 32;  // col tiles per wave
  v8f acc[PER] = {};

  for (int kc = 0; kc < K_; kc += 64) {
    __syncthreads();
    for (int i = tid; i < NC_ * 64; i += 256) {
      int n = i >> 6, k = i & 63;
      Ws[n * 64 + k] = (_Float16)W[(size_t)(kc + k) * NC_ + n];
    }
    if (kc == 0 && full) async_wait0();  // A tile landed before first use
    __syncthreads();
#pragma unroll
    for (int ks = 0; ks < 64; ks += 32) {
      v16h a = ldsA(As + (rt * 16) * K_ + kc + ks, lane, K_);
#pragma unroll
      for (int li = 0; li < PER; ++li) {
        int ct = cg + 2 * li;
        v16h b = ldsB(Ws + (ct * 16) * 64 + ks, lane, 64);
        acc[li] = wmma_f16(a, b, acc[li]);
      }
    }
  }

  const int moff = (lane < 16) ? 0 : 8;
  const int nl = lane & 15;
#pragma unroll
  for (int li = 0; li < PER; ++li) {
    int ct = cg + 2 * li;
    int col = ct * 16 + nl;
    float bv = BIAS ? bias[col] : 0.f;
#pragma unroll
    for (int i = 0; i < 8; ++i) {
      int row = row0 + rt * 16 + moff + i;
      if (row < M) {
        float v = acc[li][i] + bv;
        if (OUTH) outH[(size_t)row * NC_ + col] = (_Float16)v;
        else      outF[(size_t)row * NC_ + col] = v;
      }
    }
  }
}

// ---------------------------------------------------------------------------
// Edge delta kernel: per 64 edges: hidden = relu((pos[d]-pos[s]) @ W1 + b1)
// (K=3, VALU) then delta = hidden @ W2 + b2 (WMMA). Also emits
// h = a_dst[d] - a_src[s] + delta as f16 for the attention MLP.
// ---------------------------------------------------------------------------
__global__ __launch_bounds__(256) void edge_delta(
    const float* __restrict__ pos, const int* __restrict__ EI,
    const float* __restrict__ W1, const float* __restrict__ b1,
    const float* __restrict__ W2, const float* __restrict__ b2,
    const float* __restrict__ asrc, const float* __restrict__ adst,
    float* __restrict__ delta, _Float16* __restrict__ hbuf, int E_) {
  __shared__ _Float16 hid[64 * 64];
  __shared__ _Float16 w2t[128 * 64];
  const int tid = threadIdx.x, lane = tid & 31, w = tid >> 5;
  const int e0 = blockIdx.x * 64;

  for (int i = tid; i < 128 * 64; i += 256) {
    int n = i >> 6, k = i & 63;
    w2t[n * 64 + k] = (_Float16)W2[(size_t)k * 128 + n];
  }
  {
    int el = tid >> 2;
    int h0 = (tid & 3) * 16;
    int e = e0 + el;
    float px = 0.f, py = 0.f, pz = 0.f;
    if (e < E_) {
      int s_ = EI[e], d_ = EI[E_ + e];
      px = pos[(size_t)d_ * 3 + 0] - pos[(size_t)s_ * 3 + 0];
      py = pos[(size_t)d_ * 3 + 1] - pos[(size_t)s_ * 3 + 1];
      pz = pos[(size_t)d_ * 3 + 2] - pos[(size_t)s_ * 3 + 2];
    }
#pragma unroll
    for (int hh = 0; hh < 16; ++hh) {
      int h = h0 + hh;
      float v = px * W1[h] + py * W1[64 + h] + pz * W1[128 + h] + b1[h];
      hid[el * 64 + h] = (_Float16)fmaxf(v, 0.f);
    }
  }
  __syncthreads();

  const int rt = w & 3, cg = w >> 2;
  v8f acc[4] = {};
#pragma unroll
  for (int ks = 0; ks < 64; ks += 32) {
    v16h a = ldsA(hid + rt * 16 * 64 + ks, lane, 64);
#pragma unroll
    for (int li = 0; li < 4; ++li) {
      int ct = cg + 2 * li;
      v16h b = ldsB(w2t + ct * 16 * 64 + ks, lane, 64);
      acc[li] = wmma_f16(a, b, acc[li]);
    }
  }
  const int moff = (lane < 16) ? 0 : 8, nl = lane & 15;
#pragma unroll
  for (int li = 0; li < 4; ++li) {
    int ct = cg + 2 * li;
    int col = ct * 16 + nl;
    float bv = b2[col];
#pragma unroll
    for (int i = 0; i < 8; ++i) {
      int e = e0 + rt * 16 + moff + i;
      if (e < E_) {
        float d = acc[li][i] + bv;
        delta[(size_t)e * 128 + col] = d;
        int s_ = EI[e], dd = EI[E_ + e];
        float hv = adst[(size_t)dd * 128 + col] - asrc[(size_t)s_ * 128 + col] + d;
        hbuf[(size_t)e * 128 + col] = (_Float16)hv;
      }
    }
  }
}

// ---------------------------------------------------------------------------
// Edge attention MLP: alpha = relu(h @ W1 + b1) @ W2 + b2 (both via WMMA),
// plus atomic segment-max into mbuf[dst]. h tile staged with async copies.
// ---------------------------------------------------------------------------
__global__ __launch_bounds__(256) void edge_alpha(
    const _Float16* __restrict__ hbuf, const int* __restrict__ EI,
    const float* __restrict__ W1, const float* __restrict__ b1,
    const float* __restrict__ W2, const float* __restrict__ b2,
    float* __restrict__ alpha, float* __restrict__ mbuf, int E_) {
  __shared__ _Float16 ht[64 * 128];   // edge rows, K=128
  __shared__ _Float16 w1t[64 * 128];  // W1^T [n][k]
  __shared__ _Float16 hid[64 * 64];
  __shared__ _Float16 w2t[128 * 64];  // W2^T [n][k]
  const int tid = threadIdx.x, lane = tid & 31, w = tid >> 5;
  const int e0 = blockIdx.x * 64;
  const bool full = (e0 + 64 <= E_);

  if (full) {
    async_stage<64 * 128 * 2>(ht, hbuf + (size_t)e0 * 128, tid);
  } else {
    for (int i = tid; i < 64 * 128 / 2; i += 256) {
      int r = (2 * i) >> 7, c = (2 * i) & 127;
      unsigned v = 0;
      if (e0 + r < E_)
        v = *reinterpret_cast<const unsigned*>(hbuf + (size_t)(e0 + r) * 128 + c);
      *reinterpret_cast<unsigned*>(ht + r * 128 + c) = v;
    }
  }
  for (int i = tid; i < 64 * 128; i += 256) {
    int n = i >> 7, k = i & 127;
    w1t[n * 128 + k] = (_Float16)W1[(size_t)k * 64 + n];
  }
  for (int i = tid; i < 128 * 64; i += 256) {
    int n = i >> 6, k = i & 63;
    w2t[n * 64 + k] = (_Float16)W2[(size_t)k * 128 + n];
  }
  if (full) async_wait0();
  __syncthreads();

  const int rt = w & 3, cg = w >> 2;
  const int moff = (lane < 16) ? 0 : 8, nl = lane & 15;

  { // layer 1: 64x64 hidden, K=128
    v8f acc[2] = {};
#pragma unroll
    for (int ks = 0; ks < 128; ks += 32) {
      v16h a = ldsA(ht + rt * 16 * 128 + ks, lane, 128);
#pragma unroll
      for (int li = 0; li < 2; ++li) {
        int ct = cg + 2 * li;
        v16h b = ldsB(w1t + ct * 16 * 128 + ks, lane, 128);
        acc[li] = wmma_f16(a, b, acc[li]);
      }
    }
#pragma unroll
    for (int li = 0; li < 2; ++li) {
      int ct = cg + 2 * li;
      int col = ct * 16 + nl;
      float bv = b1[col];
#pragma unroll
      for (int i = 0; i < 8; ++i) {
        int r = rt * 16 + moff + i;
        hid[r * 64 + col] = (_Float16)fmaxf(acc[li][i] + bv, 0.f);
      }
    }
  }
  __syncthreads();

  { // layer 2: 64x128 out, K=64
    v8f acc[4] = {};
#pragma unroll
    for (int ks = 0; ks < 64; ks += 32) {
      v16h a = ldsA(hid + rt * 16 * 64 + ks, lane, 64);
#pragma unroll
      for (int li = 0; li < 4; ++li) {
        int ct = cg + 2 * li;
        v16h b = ldsB(w2t + ct * 16 * 64 + ks, lane, 64);
        acc[li] = wmma_f16(a, b, acc[li]);
      }
    }
#pragma unroll
    for (int li = 0; li < 4; ++li) {
      int ct = cg + 2 * li;
      int col = ct * 16 + nl;
      float bv = b2[col];
#pragma unroll
      for (int i = 0; i < 8; ++i) {
        int e = e0 + rt * 16 + moff + i;
        if (e < E_) {
          float av = acc[li][i] + bv;
          alpha[(size_t)e * 128 + col] = av;
          int dd = EI[E_ + e];
          atomicMaxF(&mbuf[(size_t)dd * 128 + col], av);
        }
      }
    }
  }
}

// ---------------------------------------------------------------------------
// Softmax pass 2: e = exp(alpha - m[dst]); s[dst] += e (atomic)
// ---------------------------------------------------------------------------
__global__ void edge_exp(float* __restrict__ alpha, const float* __restrict__ mbuf,
                         float* __restrict__ sbuf, const int* __restrict__ EI,
                         long total, int E_) {
  long i = (long)blockIdx.x * blockDim.x + threadIdx.x;
  if (i >= total) return;
  int e = (int)(i >> 7), c = (int)(i & 127);
  int d_ = EI[E_ + e];
  float ev = __expf(alpha[i] - mbuf[(size_t)d_ * 128 + c]);
  alpha[i] = ev;
  atomicAdd(&sbuf[(size_t)d_ * 128 + c], ev);
}

// ---------------------------------------------------------------------------
// Softmax pass 3: out[dst] += e/(s[dst]+eps) * (x_j[src] + delta)
// ---------------------------------------------------------------------------
__global__ void edge_out(const float* __restrict__ alpha, const float* __restrict__ sbuf,
                         const float* __restrict__ xj, const float* __restrict__ delta,
                         const int* __restrict__ EI, float* __restrict__ out,
                         long total, int E_) {
  long i = (long)blockIdx.x * blockDim.x + threadIdx.x;
  if (i >= total) return;
  int e = (int)(i >> 7), c = (int)(i & 127);
  int s_ = EI[e], d_ = EI[E_ + e];
  float a = alpha[i] / (sbuf[(size_t)d_ * 128 + c] + 1e-16f);
  atomicAdd(&out[(size_t)d_ * 128 + c],
            a * (xj[(size_t)s_ * 128 + c] + delta[i]));
}

// ---------------------------------------------------------------------------
extern "C" void kernel_launch(void* const* d_in, const int* in_sizes, int n_in,
                              void* d_out, int out_size, void* d_ws, size_t ws_size,
                              hipStream_t stream) {
  (void)n_in; (void)out_size; (void)ws_size;
  const float* pf        = (const float*)d_in[0];
  const float* pos       = (const float*)d_in[1];
  const float* naip      = (const float*)d_in[2];
  const float* uav       = (const float*)d_in[3];
  const float* nb        = (const float*)d_in[4];
  const float* ub        = (const float*)d_in[5];
  const float* center    = (const float*)d_in[6];
  const float* scale     = (const float*)d_in[7];
  const float* pre_W     = (const float*)d_in[8];
  const float* pre_b     = (const float*)d_in[9];
  const float* pos_W1    = (const float*)d_in[10];
  const float* pos_b1    = (const float*)d_in[11];
  const float* pos_W2    = (const float*)d_in[12];
  const float* pos_b2    = (const float*)d_in[13];
  const float* attn_W1   = (const float*)d_in[14];
  const float* attn_b1   = (const float*)d_in[15];
  const float* attn_W2   = (const float*)d_in[16];
  const float* attn_b2   = (const float*)d_in[17];
  const float* lin_W     = (const float*)d_in[18];
  const float* lin_src_W = (const float*)d_in[19];
  const float* lin_dst_W = (const float*)d_in[20];
  const int*   EI        = (const int*)d_in[21];
  float* out = (float*)d_out;

  const int Npts = in_sizes[0] / 128;
  const int E_   = in_sizes[21] / 2;

  char* p = (char*)d_ws;
  auto take = [&](size_t bytes) -> char* {
    char* r = p;
    p += (bytes + 255) & ~(size_t)255;
    return r;
  };
  _Float16* xcat = (_Float16*)take((size_t)Npts * 192 * 2);
  _Float16* xh   = (_Float16*)take((size_t)Npts * 192 * 2);
  float* asrc  = (float*)take((size_t)Npts * 128 * 4);
  float* adst  = (float*)take((size_t)Npts * 128 * 4);
  float* xj    = (float*)take((size_t)Npts * 128 * 4);
  float* mbuf  = (float*)take((size_t)Npts * 128 * 4);
  float* sbuf  = (float*)take((size_t)Npts * 128 * 4);
  float* delta = (float*)take((size_t)E_ * 128 * 4);
  float* alpha = (float*)take((size_t)E_ * 128 * 4);
  _Float16* hbuf = (_Float16*)take((size_t)E_ * 128 * 2);

  long n128 = (long)Npts * 128;
  init_bufs<<<(int)((n128 + 255) / 256), 256, 0, stream>>>(out, mbuf, sbuf, n128);

  long n192 = (long)Npts * 192;
  build_xcat<<<(int)((n192 + 255) / 256), 256, 0, stream>>>(
      pf, pos, naip, uav, nb, ub, center, scale, xcat, Npts);

  int nblk = (Npts + 63) / 64;
  gemm_node<192, 192, true, true><<<nblk, 256, (64 * 192 + 192 * 64) * 2, stream>>>(
      xcat, pre_W, pre_b, (float*)nullptr, xh, Npts);
  gemm_node<192, 128, false, false><<<nblk, 256, (64 * 192 + 128 * 64) * 2, stream>>>(
      xh, lin_src_W, (const float*)nullptr, asrc, (_Float16*)nullptr, Npts);
  gemm_node<192, 128, false, false><<<nblk, 256, (64 * 192 + 128 * 64) * 2, stream>>>(
      xh, lin_dst_W, (const float*)nullptr, adst, (_Float16*)nullptr, Npts);
  gemm_node<192, 128, false, false><<<nblk, 256, (64 * 192 + 128 * 64) * 2, stream>>>(
      xh, lin_W, (const float*)nullptr, xj, (_Float16*)nullptr, Npts);

  int eblk = (E_ + 63) / 64;
  edge_delta<<<eblk, 256, 0, stream>>>(pos, EI, pos_W1, pos_b1, pos_W2, pos_b2,
                                       asrc, adst, delta, hbuf, E_);
  edge_alpha<<<eblk, 256, 0, stream>>>(hbuf, EI, attn_W1, attn_b1, attn_W2, attn_b2,
                                       alpha, mbuf, E_);

  long e128 = (long)E_ * 128;
  edge_exp<<<(int)((e128 + 255) / 256), 256, 0, stream>>>(alpha, mbuf, sbuf, EI, e128, E_);
  edge_out<<<(int)((e128 + 255) / 256), 256, 0, stream>>>(alpha, sbuf, xj, delta, EI, out,
                                                          e128, E_);
}
